// Taco_58136677319225
// MI455X (gfx1250) — matrix operand order repaired and checked by reading, not compiled, not run on hardware
//
#include <hip/hip_runtime.h>
#include <hip/hip_bf16.h>
#include <stdint.h>

#define TT 2048
#define CC 768
#define KNN 5
#define LDSS 40   // ushort (bf16) stride per LDS row = 80 bytes (16B aligned chunks)

typedef __attribute__((ext_vector_type(16))) __bf16 v16bf;
typedef __attribute__((ext_vector_type(8)))  float  v8f;

#if defined(__has_builtin)
#if __has_builtin(__builtin_amdgcn_global_load_async_to_lds_b128) && \
    __has_builtin(__builtin_amdgcn_s_wait_asynccnt)
#define HAVE_ASYNC_LDS 1
#endif
#endif
#ifndef HAVE_ASYNC_LDS
#define HAVE_ASYNC_LDS 0
#endif

#if HAVE_ASYNC_LDS
// builtin signature (from hipcc diagnostic): (v4i AS1*, v4i AS3*, imm int, imm int)
typedef int v4i_vs __attribute__((vector_size(4 * sizeof(int))));
typedef __attribute__((address_space(1))) v4i_vs* as1_v4i_p;
typedef __attribute__((address_space(3))) v4i_vs* as3_v4i_p;
__device__ __forceinline__ as1_v4i_p as_g(const void* p) {
  return (as1_v4i_p)(unsigned long long)(size_t)p;   // global addr = generic addr
}
__device__ __forceinline__ as3_v4i_p as_l(void* p) {
  return (as3_v4i_p)(unsigned)(size_t)p;             // LDS offset = low 32 bits (aperture rule)
}
#endif

// ---------------- helpers ----------------

__device__ __forceinline__ unsigned hash3(unsigned a, unsigned b, unsigned c) {
  unsigned h = a * 0x9E3779B9u ^ (b + 0x85EBCA6Bu) * 0xC2B2AE35u ^ (c * 0x27D4EB2Fu);
  h ^= h >> 16; h *= 0x7FEB352Du; h ^= h >> 15; h *= 0x846CA68Bu; h ^= h >> 16;
  return h;
}

// round-to-nearest-even f32 -> bf16, packed pair
__device__ __forceinline__ unsigned pack_bf16x2(float x, float y) {
  unsigned bx = __float_as_uint(x), by = __float_as_uint(y);
  bx = (bx + 0x7FFFu + ((bx >> 16) & 1u)) >> 16;
  by = (by + 0x7FFFu + ((by >> 16) & 1u)) >> 16;
  return (bx & 0xFFFFu) | (by << 16);
}

union FragU { uint4 u[2]; v16bf v; };

// A-matrix 16x32 bf16 fragment: lanes 0-15 kb=0, lanes 16-31 kb=8,
// elements [0..7]=K kb..kb+7, [8..15]=K kb+16..kb+23  (ISA 7.12.2)
__device__ __forceinline__ v16bf load_fragA(const unsigned short* rowp, int lane) {
  const int kb = (lane & 16) ? 8 : 0;
  FragU f;
  f.u[0] = *(const uint4*)(rowp + kb);
  f.u[1] = *(const uint4*)(rowp + kb + 16);
  return f.v;
}
// B-matrix 32x16 bf16 fragment: lanes 0-15 kb=0, lanes 16-31 kb=8,
// elements [0..7]=K kb..kb+7, [8..15]=K kb+16..kb+23 (same as A)
__device__ __forceinline__ v16bf load_fragB(const unsigned short* rowp, int lane) {
  const int kh = (lane & 16) ? 8 : 0;
  FragU f;
  f.u[0] = *(const uint4*)(rowp + kh);
  f.u[1] = *(const uint4*)(rowp + kh + 16);
  return f.v;
}

// ---------------- f32 -> bf16 matrix conversion (done once per matrix) ----------------

__global__ __launch_bounds__(256) void to_bf16(const float* __restrict__ src,
                                               unsigned short* __restrict__ dst, int n8) {
  int i = blockIdx.x * blockDim.x + threadIdx.x;  // one thread per 8 elements
  if (i >= n8) return;
  const float4* s = (const float4*)(src + (size_t)i * 8);
  float4 a = s[0], b = s[1];
  uint4 o;
  o.x = pack_bf16x2(a.x, a.y); o.y = pack_bf16x2(a.z, a.w);
  o.z = pack_bf16x2(b.x, b.y); o.w = pack_bf16x2(b.z, b.w);
  *(uint4*)(dst + (size_t)i * 8) = o;
}

// ---------------- GEMM: G[i][j] = dot(A_i, B_j), 2048x2048, Kdim=768, bf16 operands ----

__global__ __launch_bounds__(256) void gemm_nt_bf16(const unsigned short* __restrict__ A,
                                                    const unsigned short* __restrict__ B,
                                                    float* __restrict__ G) {
  __shared__ __align__(16) unsigned short lds[2][2][128 * LDSS];  // [buf][matA/matB]
  const int tid  = threadIdx.x;
  const int lane = tid & 31;
  const int wave = tid >> 5;
  const int waveM = wave >> 2;   // 0..1 -> 64 rows each
  const int waveN = wave & 3;    // 0..3 -> 32 cols each
  const int blockM = blockIdx.x * 128;
  const int blockN = blockIdx.y * 128;

  v8f acc[4][2] = {};

  const int ldRow = tid >> 1;          // 0..127
  const int ldSeg = (tid & 1) * 16;    // bf16 element offset within 32-wide k tile
  const unsigned short* gA = A + (size_t)(blockM + ldRow) * CC + ldSeg;
  const unsigned short* gB = B + (size_t)(blockN + ldRow) * CC + ldSeg;
  const int laneN  = lane & 15;
  const int laneHi = (lane >> 4) & 1;

  auto issue_copy = [&](int kt, int buf) {
    const unsigned short* pa = gA + kt * 32;
    const unsigned short* pb = gB + kt * 32;
    unsigned short* la = &lds[buf][0][ldRow * LDSS + ldSeg];
    unsigned short* lb = &lds[buf][1][ldRow * LDSS + ldSeg];
#if HAVE_ASYNC_LDS
    __builtin_amdgcn_global_load_async_to_lds_b128(as_g(pa),     as_l(la),     0, 0);
    __builtin_amdgcn_global_load_async_to_lds_b128(as_g(pa + 8), as_l(la + 8), 0, 0);
    __builtin_amdgcn_global_load_async_to_lds_b128(as_g(pb),     as_l(lb),     0, 0);
    __builtin_amdgcn_global_load_async_to_lds_b128(as_g(pb + 8), as_l(lb + 8), 0, 0);
#else
    __builtin_prefetch((const char*)(pa + 32 * 32), 0, 3);
    uint4 a0 = *(const uint4*)pa, a1 = *(const uint4*)(pa + 8);
    uint4 b0 = *(const uint4*)pb, b1 = *(const uint4*)(pb + 8);
    *(uint4*)la = a0; *(uint4*)(la + 8) = a1;
    *(uint4*)lb = b0; *(uint4*)(lb + 8) = b1;
#endif
  };

  constexpr int NK = CC / 32;  // 24 k-steps
  issue_copy(0, 0);
  for (int kt = 0; kt < NK; ++kt) {
    const int buf = kt & 1;
    // issue next tile's async copy into the other buffer, then wait for THIS tile
    if (kt + 1 < NK) {
      issue_copy(kt + 1, buf ^ 1);
#if HAVE_ASYNC_LDS
      __builtin_amdgcn_s_wait_asynccnt(4);   // in-order: <=4 outstanding => tile kt landed
#endif
    } else {
#if HAVE_ASYNC_LDS
      __builtin_amdgcn_s_wait_asynccnt(0);
#endif
    }
    __syncthreads();                          // tile kt visible to all waves

    v16bf bf[2];
#pragma unroll
    for (int nt = 0; nt < 2; ++nt)
      bf[nt] = load_fragB(&lds[buf][1][(waveN * 32 + nt * 16 + laneN) * LDSS], lane);
#pragma unroll
    for (int mt = 0; mt < 4; ++mt) {
      v16bf af = load_fragA(&lds[buf][0][(waveM * 64 + mt * 16 + laneN) * LDSS], lane);
#pragma unroll
      for (int nt = 0; nt < 2; ++nt)
        acc[mt][nt] = __builtin_amdgcn_wmma_f32_16x16x32_bf16(
            false, af, false, bf[nt], (short)0, acc[mt][nt], false, false);
    }
    __syncthreads();                          // all readers of buf done before it is reused
  }

#pragma unroll
  for (int mt = 0; mt < 4; ++mt) {
#pragma unroll
    for (int nt = 0; nt < 2; ++nt) {
      const int row0 = blockM + waveM * 64 + mt * 16 + laneHi * 8;
      const int col  = blockN + waveN * 32 + nt * 16 + laneN;
#pragma unroll
      for (int r = 0; r < 8; ++r)
        G[(size_t)(row0 + r) * TT + col] = acc[mt][nt][r];
    }
  }
}

// ---------------- small kernels ----------------

__global__ __launch_bounds__(256) void normalize_rows(const float* __restrict__ z,
                                                      float* __restrict__ zn,
                                                      float* __restrict__ sqn) {
  const int row = blockIdx.x;
  const float* p = z + (size_t)row * CC;
  float ss = 0.f;
  for (int c = threadIdx.x; c < CC; c += 256) { float v = p[c]; ss += v * v; }
  for (int off = 16; off > 0; off >>= 1) ss += __shfl_down(ss, off, 32);
  __shared__ float red[8];
  if ((threadIdx.x & 31) == 0) red[threadIdx.x >> 5] = ss;
  __syncthreads();
  __shared__ float s_inv;
  if (threadIdx.x == 0) {
    float t = 0.f;
    for (int i = 0; i < 8; ++i) t += red[i];
    sqn[row] = t;
    s_inv = rsqrtf(t + 1e-12f);
  }
  __syncthreads();
  const float inv = s_inv;
  for (int c = threadIdx.x; c < CC; c += 256) zn[(size_t)row * CC + c] = p[c] * inv;
}

__global__ void fill_f32(float* p, int n, float v) {
  int i = blockIdx.x * blockDim.x + threadIdx.x;
  if (i < n) p[i] = v;
}
__global__ void fill_i32(int* p, int n, int v) {
  int i = blockIdx.x * blockDim.x + threadIdx.x;
  if (i < n) p[i] = v;
}

// 5 smallest keys per row; key = sign*G[i][j] + (colbias ? colbias[j] : 0), excluding j==i
__global__ __launch_bounds__(256) void topk5(const float* __restrict__ G,
                                             const float* __restrict__ colbias,
                                             float sign, int* __restrict__ outIdx) {
  int i = blockIdx.x * blockDim.x + threadIdx.x;
  if (i >= TT) return;
  float bk[KNN]; int bi[KNN];
#pragma unroll
  for (int k = 0; k < KNN; ++k) { bk[k] = 3.0e38f; bi[k] = 0; }
  const float* row = G + (size_t)i * TT;
  for (int j = 0; j < TT; ++j) {
    if (j == i) continue;
    float key = sign * row[j] + (colbias ? colbias[j] : 0.f);
    if (key < bk[KNN - 1]) {
      int k = KNN - 1;
      while (k > 0 && bk[k - 1] > key) { bk[k] = bk[k - 1]; bi[k] = bi[k - 1]; --k; }
      bk[k] = key; bi[k] = j;
    }
  }
#pragma unroll
  for (int k = 0; k < KNN; ++k) outIdx[i * KNN + k] = bi[k];
}

// 5 random non-neighbor, non-self indices per row (top-5 of hashed scores)
__global__ __launch_bounds__(256) void pick_negs(const int* __restrict__ neigh,
                                                 int* __restrict__ negs, unsigned seed) {
  int i = blockIdx.x * blockDim.x + threadIdx.x;
  if (i >= TT) return;
  int nb[KNN];
#pragma unroll
  for (int k = 0; k < KNN; ++k) nb[k] = neigh[i * KNN + k];
  unsigned bs[KNN]; int bi[KNN];
#pragma unroll
  for (int k = 0; k < KNN; ++k) { bs[k] = 0u; bi[k] = 0; }
  for (int j = 0; j < TT; ++j) {
    if (j == i) continue;
    bool isnb = false;
#pragma unroll
    for (int k = 0; k < KNN; ++k) isnb |= (nb[k] == j);
    if (isnb) continue;
    unsigned s = hash3(seed, (unsigned)i, (unsigned)j);
    if (s > bs[KNN - 1]) {
      int k = KNN - 1;
      while (k > 0 && bs[k - 1] < s) { bs[k] = bs[k - 1]; bi[k] = bi[k - 1]; --k; }
      bs[k] = s; bi[k] = j;
    }
  }
#pragma unroll
  for (int k = 0; k < KNN; ++k) negs[i * KNN + k] = bi[k];
}

// crossbrain hinge: sum over rows,k of relu(dot_neg - dot_pos + 0.05); one wave per row
__global__ __launch_bounds__(256) void cb_hinge(const float* __restrict__ zn,
                                                const int* __restrict__ neigh,
                                                const int* __restrict__ negs,
                                                float* __restrict__ sumv) {
  const int wave = threadIdx.x >> 5, lane = threadIdx.x & 31;
  const int i = blockIdx.x * 8 + wave;
  const float* a = zn + (size_t)i * CC;
  float dots[2 * KNN];
#pragma unroll
  for (int t = 0; t < 2 * KNN; ++t) {
    const int tgt = (t < KNN) ? neigh[i * KNN + t] : negs[i * KNN + (t - KNN)];
    const float* b = zn + (size_t)tgt * CC;
    float s = 0.f;
    for (int c = lane; c < CC; c += 32) s += a[c] * b[c];
    for (int off = 16; off > 0; off >>= 1) s += __shfl_xor(s, off, 32);
    dots[t] = s;
  }
  if (lane == 0) {
    float h = 0.f;
#pragma unroll
    for (int k = 0; k < KNN; ++k) {
      float v = dots[KNN + k] - dots[k] + 0.05f;  // (1-dp)-(1-dn)+m = dn_dot - dp_dot + m
      h += (v > 0.f) ? v : 0.f;
    }
    atomicAdd(sumv, h);
  }
}

__global__ __launch_bounds__(256) void argmax_row(const float* __restrict__ G, int* __restrict__ out) {
  int i = blockIdx.x * blockDim.x + threadIdx.x;
  if (i >= TT) return;
  const float* row = G + (size_t)i * TT;
  float best = -3.0e38f; int bj = 0;
  for (int j = 0; j < TT; ++j) { float v = row[j]; if (v > best) { best = v; bj = j; } }
  out[i] = bj;
}
__global__ __launch_bounds__(256) void argmax_col(const float* __restrict__ G, int* __restrict__ out) {
  int j = blockIdx.x * blockDim.x + threadIdx.x;
  if (j >= TT) return;
  float best = -3.0e38f; int bi = 0;
  for (int i = 0; i < TT; ++i) { float v = G[(size_t)i * TT + j]; if (v > best) { best = v; bi = i; } }
  out[j] = bi;
}

__global__ __launch_bounds__(256) void mutual_k(const int* __restrict__ bofa,
                                                const int* __restrict__ aofb,
                                                int* __restrict__ idxB,
                                                int* __restrict__ matched) {
  int i = blockIdx.x * blockDim.x + threadIdx.x;
  if (i >= TT) return;
  int b = bofa[i];
  bool mut = (aofb[b] == i);
  idxB[i] = mut ? b : -1;
  if (mut) atomicAdd(&matched[b], 1);
}

// order = unmatched indices first then matched (stable); negcnt = max(#unmatched, 1)
__global__ void order_k(const int* __restrict__ matched, int* __restrict__ order,
                        int* __restrict__ negcnt) {
  if (threadIdx.x == 0 && blockIdx.x == 0) {
    int c = 0;
    for (int j = 0; j < TT; ++j) if (matched[j] == 0) order[c++] = j;
    int unm = c;
    for (int j = 0; j < TT; ++j) if (matched[j] != 0) order[c++] = j;
    negcnt[0] = unm > 0 ? unm : 1;
  }
}

__global__ __launch_bounds__(256) void nrc_hinge(const float* __restrict__ G,
                                                 const int* __restrict__ neigh,
                                                 const int* __restrict__ idxB,
                                                 const int* __restrict__ order,
                                                 const int* __restrict__ negcnt,
                                                 unsigned seed,
                                                 float* __restrict__ sumv,
                                                 float* __restrict__ cntv) {
  int i = blockIdx.x * blockDim.x + threadIdx.x;
  if (i >= TT) return;
  const int ib = idxB[i];
  const int uB = (ib < 0) ? 0 : ib;
  const int nc = negcnt[0];
  float hs = 0.f; int np = 0;
#pragma unroll
  for (int k = 0; k < KNN; ++k) {
    int nbr = neigh[i * KNN + k];
    int pB = idxB[nbr];
    bool pm = (pB >= 0);
    int pidx = pm ? pB : 0;
    float dpos = 1.f - G[(size_t)uB * TT + pidx];
    int r = (int)(hash3(seed, (unsigned)i, (unsigned)k) % (unsigned)TT);
    int nB = order[r % nc];
    float dneg = 1.f - G[(size_t)uB * TT + nB];
    float hv = dpos - dneg + 0.4f;
    hv = (hv > 0.f) ? hv : 0.f;
    if (pm) { hs += hv; ++np; }
  }
  float per = hs / (float)(np > 0 ? np : 1);
  bool valid = (ib >= 0) && (np > 0);
  if (valid) { atomicAdd(sumv, per); atomicAdd(cntv, 1.f); }
}

__global__ void finalize(const float* __restrict__ sumv, const float* __restrict__ cntv,
                         float scale, int useCnt, float* __restrict__ out) {
  if (threadIdx.x == 0 && blockIdx.x == 0) {
    float s = sumv[0];
    if (useCnt) { float c = cntv[0]; s = (c > 0.f) ? s / c : 0.f; }
    atomicAdd(out, s * scale);
  }
}

// ---------------- driver ----------------

extern "C" void kernel_launch(void* const* d_in, const int* in_sizes, int n_in,
                              void* d_out, int out_size, void* d_ws, size_t ws_size,
                              hipStream_t stream) {
  (void)in_sizes; (void)n_in; (void)out_size; (void)ws_size;
  const float* z1 = (const float*)d_in[0];  // (4, 2048, 768) f32
  const float* z2 = (const float*)d_in[1];
  const size_t TC = (size_t)TT * CC;

  char* w = (char*)d_ws;
  float* z1n   = (float*)w; w += 4 * TC * sizeof(float);
  float* z2n   = (float*)w; w += 4 * TC * sizeof(float);
  float* G     = (float*)w; w += (size_t)TT * TT * sizeof(float);
  unsigned short* z1b  = (unsigned short*)w; w += 4 * TC * sizeof(unsigned short);
  unsigned short* z2b  = (unsigned short*)w; w += 4 * TC * sizeof(unsigned short);
  unsigned short* z1nb = (unsigned short*)w; w += 4 * TC * sizeof(unsigned short);
  unsigned short* z2nb = (unsigned short*)w; w += 4 * TC * sizeof(unsigned short);
  float* sq1   = (float*)w; w += 4 * TT * sizeof(float);
  float* sq2   = (float*)w; w += 4 * TT * sizeof(float);
  int*   neigh = (int*)w;   w += TT * KNN * sizeof(int);
  int*   negs  = (int*)w;   w += TT * KNN * sizeof(int);
  int*   bofa  = (int*)w;   w += TT * sizeof(int);
  int*   aofb  = (int*)w;   w += TT * sizeof(int);
  int*   idxB  = (int*)w;   w += TT * sizeof(int);
  int*   matched = (int*)w; w += TT * sizeof(int);
  int*   order = (int*)w;   w += TT * sizeof(int);
  int*   negcnt = (int*)w;  w += sizeof(int);
  float* sumv  = (float*)w; w += sizeof(float);
  float* cntv  = (float*)w; w += sizeof(float);

  const dim3 gg(16, 16, 1);
  const int n8 = (int)(4 * TC / 8);  // elements/8 per matrix

  normalize_rows<<<4 * TT, 256, 0, stream>>>(z1, z1n, sq1);
  normalize_rows<<<4 * TT, 256, 0, stream>>>(z2, z2n, sq2);
  to_bf16<<<n8 / 256, 256, 0, stream>>>(z1,  z1b,  n8);
  to_bf16<<<n8 / 256, 256, 0, stream>>>(z2,  z2b,  n8);
  to_bf16<<<n8 / 256, 256, 0, stream>>>(z1n, z1nb, n8);
  to_bf16<<<n8 / 256, 256, 0, stream>>>(z2n, z2nb, n8);
  fill_f32<<<1, 32, 0, stream>>>((float*)d_out, 1, 0.f);

  const float cbScale = 10.f / (4.f * (float)TT * (float)KNN);
  for (int b = 0; b < 4; ++b) {
    // crossbrain(z1[b], z2[b]): neighbors from sq_dist(z1[b]) -> min(sqn_j - 2*G)
    gemm_nt_bf16<<<gg, 256, 0, stream>>>(z1b + b * TC, z1b + b * TC, G);
    topk5<<<TT / 256, 256, 0, stream>>>(G, sq1 + b * TT, -2.0f, neigh);
    pick_negs<<<TT / 256, 256, 0, stream>>>(neigh, negs, 0x1000u + (unsigned)b);
    fill_f32<<<1, 32, 0, stream>>>(sumv, 1, 0.f);
    cb_hinge<<<TT / 8, 256, 0, stream>>>(z2n + b * TC, neigh, negs, sumv);
    finalize<<<1, 1, 0, stream>>>(sumv, cntv, cbScale, 0, (float*)d_out);

    // crossbrain(z2[b], z1[b])
    gemm_nt_bf16<<<gg, 256, 0, stream>>>(z2b + b * TC, z2b + b * TC, G);
    topk5<<<TT / 256, 256, 0, stream>>>(G, sq2 + b * TT, -2.0f, neigh);
    pick_negs<<<TT / 256, 256, 0, stream>>>(neigh, negs, 0x2000u + (unsigned)b);
    fill_f32<<<1, 32, 0, stream>>>(sumv, 1, 0.f);
    cb_hinge<<<TT / 8, 256, 0, stream>>>(z1n + b * TC, neigh, negs, sumv);
    finalize<<<1, 1, 0, stream>>>(sumv, cntv, cbScale, 0, (float*)d_out);
  }

  static const int prs[6][2] = {{0, 1}, {0, 2}, {1, 3}, {2, 0}, {3, 1}, {3, 2}};
  const float nrcScale = 10.f / 6.f;
  for (int d = 0; d < 2; ++d) {
    const unsigned short* An = d ? z2nb : z1nb;
    const unsigned short* Bn = d ? z1nb : z2nb;
    for (int t = 0; t < 6; ++t) {
      const unsigned short* A = An + (size_t)prs[t][0] * TC;
      const unsigned short* B = Bn + (size_t)prs[t][1] * TC;
      gemm_nt_bf16<<<gg, 256, 0, stream>>>(A, B, G);            // cos-sim(A,B)
      argmax_row<<<TT / 256, 256, 0, stream>>>(G, bofa);
      argmax_col<<<TT / 256, 256, 0, stream>>>(G, aofb);
      fill_i32<<<TT / 256, 256, 0, stream>>>(matched, TT, 0);
      mutual_k<<<TT / 256, 256, 0, stream>>>(bofa, aofb, idxB, matched);
      gemm_nt_bf16<<<gg, 256, 0, stream>>>(A, A, G);            // cos-sim(A,A)
      topk5<<<TT / 256, 256, 0, stream>>>(G, (const float*)nullptr, -1.0f, neigh);
      order_k<<<1, 1, 0, stream>>>(matched, order, negcnt);
      gemm_nt_bf16<<<gg, 256, 0, stream>>>(B, B, G);            // cos-sim(B,B) = DB source
      fill_f32<<<1, 32, 0, stream>>>(sumv, 2, 0.f);             // clears sumv + cntv
      nrc_hinge<<<TT / 256, 256, 0, stream>>>(G, neigh, idxB, order, negcnt,
                                              0x3000u + (unsigned)(d * 16 + t), sumv, cntv);
      finalize<<<1, 1, 0, stream>>>(sumv, cntv, nrcScale, 1, (float*)d_out);
    }
  }
}
